// SimpleGCNWithStatic_45019847197234
// MI455X (gfx1250) — matrix operand order repaired and checked by reading, not compiled, not run on hardware
//
#include <hip/hip_runtime.h>
#include <hip/hip_bf16.h>

typedef __attribute__((ext_vector_type(16))) _Float16 v16h;
typedef __attribute__((ext_vector_type(8)))  float    v8f;
typedef __attribute__((ext_vector_type(4)))  float    v4f;

#define N_  20000
#define E_  640000
#define TD_ 112
#define SD_ 64
#define SP_ 16
#define CD_ 128
#define H_  128

// -------------------- degree / norm kernels --------------------

__global__ void deg_init_kernel(float* dinv) {
    int i = blockIdx.x * blockDim.x + threadIdx.x;
    if (i < N_) dinv[i] = 1.0f;   // self-loop contributes 1
}

__global__ void deg_edge_kernel(const long long* __restrict__ ei, float* dinv) {
    int e = blockIdx.x * blockDim.x + threadIdx.x;
    if (e < E_) atomicAdd(&dinv[(int)ei[e]], 1.0f);   // row = ei[0][e]
}

__global__ void deg_fin_kernel(float* dinv) {
    int i = blockIdx.x * blockDim.x + threadIdx.x;
    if (i < N_) dinv[i] = rsqrtf(dinv[i]);            // deg >= 1 always
}

// -------------------- feature assembly --------------------

__global__ void copy_temporal_kernel(const float* __restrict__ t, float* __restrict__ x0) {
    int i = blockIdx.x * blockDim.x + threadIdx.x;
    if (i < N_ * TD_) {
        int node = i / TD_;
        int j    = i % TD_;
        x0[node * CD_ + j] = t[i];
    }
}

// -------------------- WMMA GEMM: Y[:, ycol0+...] = relu?(X@W + b) --------------------
// One wave per 16x16 output tile. fp32 accuracy via hi/lo f16 split (3 WMMAs / K-step).
// Fragments are loaded DIRECTLY from global memory in ISA 7.12.2 layout:
//   A (16-bit 16x32): lane l -> row m=l&15; K = 8*(l>>4)+{0..7} and 16+8*(l>>4)+{0..7}
//     => two contiguous 8-float chunks => global_load_b128 x4, no LDS.
//   B (16-bit 32x16): lane l -> col n=l&15; K = 8*(l>>4)+{0..7} and 16+8*(l>>4)+{0..7}
//     => ldw-strided scalar loads, coalesced across lanes.
//   C/D: lane l -> N=l&15; VGPR v -> M = v + 8*(l>>4).

__global__ void __launch_bounds__(32)
wmma_gemm_kernel(const float* __restrict__ X, int ldx,
                 const float* __restrict__ W, int ldw,
                 const float* __restrict__ bias,
                 float* __restrict__ Y, int ldy, int ycol0,
                 int K, int do_relu)
{
    const int lane = threadIdx.x;          // 0..31
    const int row0 = blockIdx.x * 16;
    const int col0 = blockIdx.y * 16;
    const int half = lane >> 4;            // 0: lanes 0-15, 1: lanes 16-31
    const int m    = lane & 15;

    const float* Arow = X + (size_t)(row0 + m) * ldx;   // this lane's A row
    const float* Bcol = W + col0 + m;                   // this lane's B column

    v8f c = {};

    for (int k0 = 0; k0 < K; k0 += 32) {
        // ---- A fragment: K = (k0 + 8*half) + {0..7}  and  +16 + {0..7}
        const int ka = k0 + half * 8;
        v4f a0 = *(const v4f*)(Arow + ka);
        v4f a1 = *(const v4f*)(Arow + ka + 4);
        v4f a2 = *(const v4f*)(Arow + ka + 16);
        v4f a3 = *(const v4f*)(Arow + ka + 20);
        if (k0 + 32 < K) __builtin_prefetch(Arow + ka + 32, 0, 1);

        v16h ahi, alo;
        #pragma unroll
        for (int i = 0; i < 4; ++i) {
            float v;
            _Float16 hi;
            v = a0[i]; hi = (_Float16)v; ahi[i]      = hi; alo[i]      = (_Float16)(v - (float)hi);
            v = a1[i]; hi = (_Float16)v; ahi[4 + i]  = hi; alo[4 + i]  = (_Float16)(v - (float)hi);
            v = a2[i]; hi = (_Float16)v; ahi[8 + i]  = hi; alo[8 + i]  = (_Float16)(v - (float)hi);
            v = a3[i]; hi = (_Float16)v; ahi[12 + i] = hi; alo[12 + i] = (_Float16)(v - (float)hi);
        }

        // ---- B fragment: col n=m, K = (k0 + 8*half) + {0..7}  and  +16 + {0..7}
        const int kb = k0 + half * 8;
        v16h bhi, blo;
        #pragma unroll
        for (int i = 0; i < 16; ++i) {
            float v = Bcol[(size_t)(kb + i + (i >= 8 ? 8 : 0)) * ldw];
            _Float16 hi = (_Float16)v;
            bhi[i] = hi;
            blo[i] = (_Float16)(v - (float)hi);
        }

        // X@W = hi*hi + hi*lo + lo*hi  (lo*lo below fp32 noise floor)
        c = __builtin_amdgcn_wmma_f32_16x16x32_f16(false, ahi, false, bhi, (short)0, c, false, false);
        c = __builtin_amdgcn_wmma_f32_16x16x32_f16(false, ahi, false, blo, (short)0, c, false, false);
        c = __builtin_amdgcn_wmma_f32_16x16x32_f16(false, alo, false, bhi, (short)0, c, false, false);
    }

    // Epilogue: lane l -> N = l&15, VGPR v -> M = v + 8*(l>>4)
    const int n    = m;
    const int mrow = half * 8;
    const float bv = bias ? bias[col0 + n] : 0.0f;
    #pragma unroll
    for (int v = 0; v < 8; ++v) {
        float val = c[v] + bv;
        if (do_relu) val = fmaxf(val, 0.0f);
        Y[(size_t)(row0 + mrow + v) * ldy + ycol0 + col0 + n] = val;
    }
}

// -------------------- aggregation --------------------

// h = dinv^2 * xw + bias   (self loop + bias, initializes h)
__global__ void agg_self_kernel(const float* __restrict__ xw, const float* __restrict__ dinv,
                                const float* __restrict__ bias, float* __restrict__ h) {
    int i = blockIdx.x * blockDim.x + threadIdx.x;
    if (i < N_ * H_) {
        int node = i >> 7;
        int j    = i & 127;
        float d  = dinv[node];
        h[i] = d * d * xw[i] + bias[j];
    }
}

// h[row] += dinv[row]*dinv[col] * xw[col]  for every edge/feature
__global__ void agg_edge_kernel(const long long* __restrict__ ei,
                                const float* __restrict__ xw,
                                const float* __restrict__ dinv,
                                float* __restrict__ h) {
    int idx = blockIdx.x * blockDim.x + threadIdx.x;
    if (idx < E_ * H_) {
        int e = idx >> 7;
        int j = idx & 127;
        int r = (int)ei[e];
        int c = (int)ei[E_ + e];
        float norm = dinv[r] * dinv[c];
        atomicAdd(&h[(size_t)r * H_ + j], norm * xw[(size_t)c * H_ + j]);
    }
}

__global__ void relu_kernel(float* __restrict__ h) {
    int i = blockIdx.x * blockDim.x + threadIdx.x;
    if (i < N_ * H_) h[i] = fmaxf(h[i], 0.0f);
}

// -------------------- final projection: out[n] = h[n]@Wl + bl --------------------

__global__ void final_dot_kernel(const float* __restrict__ h, const float* __restrict__ Wl,
                                 const float* __restrict__ bl, float* __restrict__ out) {
    int gid  = blockIdx.x * blockDim.x + threadIdx.x;
    int node = gid >> 5;
    int lane = gid & 31;
    if (node < N_) {
        const float* row = h + (size_t)node * H_;
        float acc = 0.0f;
        #pragma unroll
        for (int i = 0; i < 4; ++i)
            acc += row[lane + 32 * i] * Wl[lane + 32 * i];
        #pragma unroll
        for (int off = 16; off > 0; off >>= 1)
            acc += __shfl_xor(acc, off, 32);
        if (lane == 0) out[node] = acc + bl[0];
    }
}

// -------------------- launcher --------------------

extern "C" void kernel_launch(void* const* d_in, const int* in_sizes, int n_in,
                              void* d_out, int out_size, void* d_ws, size_t ws_size,
                              hipStream_t stream) {
    (void)in_sizes; (void)n_in; (void)out_size; (void)ws_size;

    const float*     temporal = (const float*)d_in[0];
    const float*     statics  = (const float*)d_in[1];
    const long long* ei       = (const long long*)d_in[2];
    const float*     Ws = (const float*)d_in[3];
    const float*     bs = (const float*)d_in[4];
    const float*     W1 = (const float*)d_in[5];
    const float*     b1 = (const float*)d_in[6];
    const float*     W2 = (const float*)d_in[7];
    const float*     b2 = (const float*)d_in[8];
    const float*     Wl = (const float*)d_in[9];
    const float*     bl = (const float*)d_in[10];
    float* out = (float*)d_out;

    float* ws   = (float*)d_ws;
    float* dinv = ws;                                   // N (padded to 20480)
    float* x0   = ws + 20480;                           // N*CD
    float* xw   = x0 + (size_t)N_ * CD_;                // N*H
    float* h1   = xw + (size_t)N_ * H_;                 // N*H
    float* h2   = x0;                                   // reuse x0 after layer-1 GEMM

    // degrees + normalization
    deg_init_kernel<<<(N_ + 255) / 256, 256, 0, stream>>>(dinv);
    deg_edge_kernel<<<(E_ + 255) / 256, 256, 0, stream>>>(ei, dinv);
    deg_fin_kernel <<<(N_ + 255) / 256, 256, 0, stream>>>(dinv);

    // x0 = [temporal | relu(static@Ws + bs)]
    copy_temporal_kernel<<<(N_ * TD_ + 255) / 256, 256, 0, stream>>>(temporal, x0);
    wmma_gemm_kernel<<<dim3(N_ / 16, SP_ / 16), 32, 0, stream>>>(
        statics, SD_, Ws, SP_, bs, x0, CD_, TD_, SD_, 1);

    // layer 1: xw = x0@W1 ; h1 = relu(agg(xw) + b1)
    wmma_gemm_kernel<<<dim3(N_ / 16, H_ / 16), 32, 0, stream>>>(
        x0, CD_, W1, H_, nullptr, xw, H_, 0, CD_, 0);
    agg_self_kernel<<<(N_ * H_ + 255) / 256, 256, 0, stream>>>(xw, dinv, b1, h1);
    agg_edge_kernel<<<(E_ * H_ + 255) / 256, 256, 0, stream>>>(ei, xw, dinv, h1);
    relu_kernel    <<<(N_ * H_ + 255) / 256, 256, 0, stream>>>(h1);

    // layer 2: xw = h1@W2 ; h2 = relu(agg(xw) + b2)
    wmma_gemm_kernel<<<dim3(N_ / 16, H_ / 16), 32, 0, stream>>>(
        h1, H_, W2, H_, nullptr, xw, H_, 0, H_, 0);
    agg_self_kernel<<<(N_ * H_ + 255) / 256, 256, 0, stream>>>(xw, dinv, b2, h2);
    agg_edge_kernel<<<(E_ * H_ + 255) / 256, 256, 0, stream>>>(ei, xw, dinv, h2);
    relu_kernel    <<<(N_ * H_ + 255) / 256, 256, 0, stream>>>(h2);

    // out = h2@Wl + bl
    final_dot_kernel<<<(N_ * 32 + 255) / 256, 256, 0, stream>>>(h2, Wl, bl, out);
}
